// GraphLatentReasoning_GraphSAGE_28174985462349
// MI455X (gfx1250) — matrix operand order repaired and checked by reading, not compiled, not run on hardware
//
#include <hip/hip_runtime.h>
#include <hip/hip_bf16.h>

typedef __bf16 bf16_t;
typedef __attribute__((ext_vector_type(16))) __bf16 v16bf;
typedef __attribute__((ext_vector_type(8)))  __bf16 v8bf;
typedef __attribute__((ext_vector_type(8)))  float  v8f;

#define DFEAT 768
#define DCH   192   // DFEAT/4

// ---------------------------------------------------------------- utilities

__global__ void k_zero_f32(float* p, long long n) {
    long long i = (long long)blockIdx.x * blockDim.x + threadIdx.x;
    if (i < n) p[i] = 0.0f;
}

__global__ void k_cvt_bf16(const float* __restrict__ in, bf16_t* __restrict__ out, long long n) {
    long long i = (long long)blockIdx.x * blockDim.x + threadIdx.x;
    if (i < n) out[i] = (bf16_t)in[i];
}

// ---------------------------------------------------------------- aggregation

__global__ void k_count_edges(const int* __restrict__ ei, int E, float* __restrict__ cnt) {
    int i = blockIdx.x * blockDim.x + threadIdx.x;
    if (i < E) unsafeAtomicAdd(&cnt[ei[E + i]], 1.0f);   // dst row of edge_index
}

__global__ void k_scatter_f32(const float* __restrict__ x, const int* __restrict__ ei,
                              int E, float* __restrict__ agg) {
    long long idx = (long long)blockIdx.x * blockDim.x + threadIdx.x;
    if (idx >= (long long)E * DCH) return;
    int e = (int)(idx / DCH);
    int c = (int)(idx % DCH) * 4;
    int s = ei[e];
    int d = ei[E + e];
    const float4 v = *(const float4*)(x + (size_t)s * DFEAT + c);
    float* p = agg + (size_t)d * DFEAT + c;
    unsafeAtomicAdd(p + 0, v.x);
    unsafeAtomicAdd(p + 1, v.y);
    unsafeAtomicAdd(p + 2, v.z);
    unsafeAtomicAdd(p + 3, v.w);
}

__global__ void k_scatter_bf16(const bf16_t* __restrict__ x, const int* __restrict__ ei,
                               int E, float* __restrict__ agg) {
    long long idx = (long long)blockIdx.x * blockDim.x + threadIdx.x;
    if (idx >= (long long)E * DCH) return;
    int e = (int)(idx / DCH);
    int c = (int)(idx % DCH) * 4;
    int s = ei[e];
    int d = ei[E + e];
    const bf16_t* sp = x + (size_t)s * DFEAT + c;
    float* p = agg + (size_t)d * DFEAT + c;
#pragma unroll
    for (int j = 0; j < 4; ++j) unsafeAtomicAdd(p + j, (float)sp[j]);
}

__global__ void k_normalize_bf16(const float* __restrict__ agg, const float* __restrict__ cnt,
                                 bf16_t* __restrict__ out, long long n) {
    long long i = (long long)blockIdx.x * blockDim.x + threadIdx.x;
    if (i >= n) return;
    int node = (int)(i / DFEAT);
    float c = fmaxf(cnt[node], 1.0f);
    out[i] = (bf16_t)(agg[i] / c);
}

// ---------------------------------------------------------------- WMMA GEMM
// C[N,768] = act( A1 @ W1^T + A2 @ W2^T + bias )
// A row-major [N,768] bf16; W row-major [768,768] bf16 (B[k][n] = W[n*768+k]).
// Each wave computes a 16x64 strip (4 accumulators) so the A operand is loaded
// once per K-step and reused across 4 B tiles (cuts L2 traffic that competes
// with the atomic-heavy aggregation phase). Block = 4 waves -> 16x256 strip.
//
// A-operand per-lane layout (16-bit A, 16x32): lane half selects kbA in {0,16};
// register elems 0..15 = contiguous K kbA+0..15.
// B-operand per-lane layout (16-bit B, 32x16): col = lane&15, lane half selects
// kbB in {0,16}; elems 0..15 = contiguous K.

__device__ inline v16bf load_a_tile(const bf16_t* p) {
    v8bf lo = *(const v8bf*)(p);
    v8bf hi = *(const v8bf*)(p + 8);
    v16bf r;
#pragma unroll
    for (int i = 0; i < 8; ++i) { r[i] = lo[i]; r[i + 8] = hi[i]; }
    return r;
}

__global__ void __launch_bounds__(128)
k_gemm_dual(const bf16_t* __restrict__ A1, const bf16_t* __restrict__ W1,
            const bf16_t* __restrict__ A2, const bf16_t* __restrict__ W2,
            const float* __restrict__ bias,
            bf16_t* __restrict__ outbf, float* __restrict__ outf,
            int N, int relu) {
    const int wave = threadIdx.x >> 5;
    const int lane = threadIdx.x & 31;
    const int m0 = blockIdx.x * 16;
    // 48 n-tiles total: grid.y = 3, 4 waves/block, 4 tiles/wave
    const int n0 = (blockIdx.y * 16 + wave * 4) * 16;
    const int mr  = lane & 15;
    const int hih = lane >> 4;          // lane half
    int row = m0 + mr; if (row >= N) row = N - 1;
    const int kbA = hih * 16;
    const int kbB = hih * 16;

    const bf16_t* a1p = A1 + (size_t)row * DFEAT + kbA;
    const bf16_t* a2p = A2 + (size_t)row * DFEAT + kbA;
    const bf16_t* w1p = W1 + (size_t)(n0 + mr) * DFEAT + kbB;
    const bf16_t* w2p = W2 + (size_t)(n0 + mr) * DFEAT + kbB;

    v8f acc0 = {}, acc1 = {}, acc2 = {}, acc3 = {};
#pragma unroll 2
    for (int k0 = 0; k0 < DFEAT; k0 += 32) {
        __builtin_prefetch(a1p + k0 + 128, 0, 1);
        __builtin_prefetch(a2p + k0 + 128, 0, 1);
        const v16bf a1 = load_a_tile(a1p + k0);
        const v16bf a2 = load_a_tile(a2p + k0);

        v16bf b;
        b = *(const v16bf*)(w1p + k0 + 0 * 16 * DFEAT);
        acc0 = __builtin_amdgcn_wmma_f32_16x16x32_bf16(false, a1, false, b, (short)0, acc0, false, false);
        b = *(const v16bf*)(w2p + k0 + 0 * 16 * DFEAT);
        acc0 = __builtin_amdgcn_wmma_f32_16x16x32_bf16(false, a2, false, b, (short)0, acc0, false, false);

        b = *(const v16bf*)(w1p + k0 + 1 * 16 * DFEAT);
        acc1 = __builtin_amdgcn_wmma_f32_16x16x32_bf16(false, a1, false, b, (short)0, acc1, false, false);
        b = *(const v16bf*)(w2p + k0 + 1 * 16 * DFEAT);
        acc1 = __builtin_amdgcn_wmma_f32_16x16x32_bf16(false, a2, false, b, (short)0, acc1, false, false);

        b = *(const v16bf*)(w1p + k0 + 2 * 16 * DFEAT);
        acc2 = __builtin_amdgcn_wmma_f32_16x16x32_bf16(false, a1, false, b, (short)0, acc2, false, false);
        b = *(const v16bf*)(w2p + k0 + 2 * 16 * DFEAT);
        acc2 = __builtin_amdgcn_wmma_f32_16x16x32_bf16(false, a2, false, b, (short)0, acc2, false, false);

        b = *(const v16bf*)(w1p + k0 + 3 * 16 * DFEAT);
        acc3 = __builtin_amdgcn_wmma_f32_16x16x32_bf16(false, a1, false, b, (short)0, acc3, false, false);
        b = *(const v16bf*)(w2p + k0 + 3 * 16 * DFEAT);
        acc3 = __builtin_amdgcn_wmma_f32_16x16x32_bf16(false, a2, false, b, (short)0, acc3, false, false);
    }

    v8f accs[4] = {acc0, acc1, acc2, acc3};
#pragma unroll
    for (int j = 0; j < 4; ++j) {
        const int nj = n0 + j * 16 + mr;   // C column = lane&15 within tile j
        const float bv = bias[nj];
#pragma unroll
        for (int r = 0; r < 8; ++r) {
            int mrow = m0 + r + hih * 8;   // C row = vgpr + 8*lane_half
            float v = accs[j][r] + bv;
            if (relu) v = fmaxf(v, 0.0f);
            if (mrow < N) {
                size_t o = (size_t)mrow * DFEAT + nj;
                if (outbf) outbf[o] = (bf16_t)v;
                if (outf)  outf[o]  = v;
            }
        }
    }
}

// ---------------------------------------------------------------- reductions

__global__ void k_colmean(const float* __restrict__ e, int N, float* __restrict__ emb) {
    const int d = blockIdx.x;           // 768 blocks
    float s = 0.0f;
    for (int i = threadIdx.x; i < N; i += blockDim.x)
        s += e[(size_t)i * DFEAT + d];
    __shared__ float red[256];
    red[threadIdx.x] = s;
    __syncthreads();
    for (int w = 128; w > 0; w >>= 1) {
        if (threadIdx.x < w) red[threadIdx.x] += red[threadIdx.x + w];
        __syncthreads();
    }
    if (threadIdx.x == 0) emb[d] = red[0] / (float)N;
}

__global__ void k_copy_row(const float* __restrict__ e, const int* __restrict__ vidx,
                           float* __restrict__ out) {
    int d = blockIdx.x * blockDim.x + threadIdx.x;
    if (d < DFEAT) out[d] = e[(size_t)vidx[0] * DFEAT + d];
}

// ---------------------------------------------------------------- finalize

__global__ void __launch_bounds__(256)
k_finalize(const float* __restrict__ emb1, const float* __restrict__ emb2,
           const float* __restrict__ embt,
           const float* __restrict__ linW, const float* __restrict__ linb,
           const float* __restrict__ Wo, const float* __restrict__ ov,
           const int* __restrict__ opp, const float* __restrict__ labels,
           float* __restrict__ out) {
    __shared__ float feats[3 * DFEAT];
    __shared__ float ovv[DFEAT];
    __shared__ float red0[256], red1[256], red2[256];
    const int t = threadIdx.x;

    for (int i = t; i < DFEAT; i += 256) {
        float a = emb1[i], b = emb2[i];
        feats[i]             = a;
        feats[DFEAT + i]     = b;
        feats[2 * DFEAT + i] = a * b;
    }
    __syncthreads();

    const int op = opp[0] - 1;
    float pd = 0.0f, po = 0.0f, pt = 0.0f;
    for (int r = t; r < DFEAT; r += 256) {
        float s = linb[r];
        const float* wrow = linW + (size_t)r * (3 * DFEAT);
        for (int k = 0; k < 3 * DFEAT; ++k) s += wrow[k] * feats[k];
        float o  = s * Wo[(size_t)op * DFEAT + r];
        float tv = embt[r] + ov[(size_t)op * DFEAT + r];
        ovv[r] = o;
        pd += o * tv; po += o * o; pt += tv * tv;
    }
    red0[t] = pd; red1[t] = po; red2[t] = pt;
    __syncthreads();
    for (int w = 128; w > 0; w >>= 1) {
        if (t < w) { red0[t] += red0[t + w]; red1[t] += red1[t + w]; red2[t] += red2[t + w]; }
        __syncthreads();
    }
    if (t == 0) {
        const float eps = 1e-8f;
        float score = red0[0] / (fmaxf(sqrtf(red1[0]), eps) * fmaxf(sqrtf(red2[0]), eps));
        float lab   = labels[0];
        float diff  = score - lab;
        out[0] = diff * diff;   // loss
        out[1] = score;         // scores[0]
        out[2] = lab;           // labels[0]
    }
    __syncthreads();
    for (int r = t; r < DFEAT; r += 256) out[3 + r] = ovv[r];
}

// ---------------------------------------------------------------- host side

static inline size_t alup(size_t x) { return (x + 255) & ~(size_t)255; }
static inline unsigned nblk(long long n, int b) { return (unsigned)((n + b - 1) / b); }

static void process_graph(const float* x, const int* ei, int N, int E,
                          const bf16_t* wl1, const bf16_t* wr1, const float* bl1,
                          const bf16_t* wl2, const bf16_t* wr2, const float* bl2,
                          bf16_t* xbf, float* f32buf, bf16_t* aggbf, bf16_t* hbf, float* cnt,
                          float* emb, const int* vidx, float* rowout, hipStream_t stream) {
    const long long ND = (long long)N * DFEAT;
    dim3 gemmGrid(nblk(N, 16), 3);      // 48 n-tiles = 3 blocks * 4 waves * 4 tiles

    // x -> bf16
    k_cvt_bf16<<<nblk(ND, 256), 256, 0, stream>>>(x, xbf, ND);
    // degree counts (shared by both layers)
    k_zero_f32<<<nblk(N, 256), 256, 0, stream>>>(cnt, N);
    k_count_edges<<<nblk(E, 256), 256, 0, stream>>>(ei, E, cnt);

    // --- layer 1: agg = mean_{j->i} x_j ; h = relu(agg@Wl1^T + x@Wr1^T + bl1)
    k_zero_f32<<<nblk(ND, 256), 256, 0, stream>>>(f32buf, ND);
    k_scatter_f32<<<nblk((long long)E * DCH, 256), 256, 0, stream>>>(x, ei, E, f32buf);
    k_normalize_bf16<<<nblk(ND, 256), 256, 0, stream>>>(f32buf, cnt, aggbf, ND);
    k_gemm_dual<<<gemmGrid, 128, 0, stream>>>(aggbf, wl1, xbf, wr1, bl1,
                                              hbf, (float*)nullptr, N, /*relu=*/1);

    // --- layer 2: e = agg(h)@Wl2^T + h@Wr2^T + bl2  (f32 output into f32buf)
    k_zero_f32<<<nblk(ND, 256), 256, 0, stream>>>(f32buf, ND);
    k_scatter_bf16<<<nblk((long long)E * DCH, 256), 256, 0, stream>>>(hbf, ei, E, f32buf);
    k_normalize_bf16<<<nblk(ND, 256), 256, 0, stream>>>(f32buf, cnt, aggbf, ND);
    k_gemm_dual<<<gemmGrid, 128, 0, stream>>>(aggbf, wl2, hbf, wr2, bl2,
                                              (bf16_t*)nullptr, f32buf, N, /*relu=*/0);

    // --- readout
    k_colmean<<<DFEAT, 256, 0, stream>>>(f32buf, N, emb);
    if (rowout) k_copy_row<<<3, 256, 0, stream>>>(f32buf, vidx, rowout);
}

extern "C" void kernel_launch(void* const* d_in, const int* in_sizes, int n_in,
                              void* d_out, int out_size, void* d_ws, size_t ws_size,
                              hipStream_t stream) {
    const float* x1   = (const float*)d_in[0];
    const int*   ei1  = (const int*)  d_in[1];
    const int*   vidx = (const int*)  d_in[2];
    const float* xt   = (const float*)d_in[3];
    const int*   eit  = (const int*)  d_in[4];
    const int*   opn  = (const int*)  d_in[5];
    const float* lab  = (const float*)d_in[6];
    const float* Wl1  = (const float*)d_in[7];
    const float* bl1  = (const float*)d_in[8];
    const float* Wr1  = (const float*)d_in[9];
    const float* Wl2  = (const float*)d_in[10];
    const float* bl2  = (const float*)d_in[11];
    const float* Wr2  = (const float*)d_in[12];
    const float* linW = (const float*)d_in[13];
    const float* linb = (const float*)d_in[14];
    const float* Wo   = (const float*)d_in[15];
    const float* ov   = (const float*)d_in[16];

    const int N1 = in_sizes[0] / DFEAT;
    const int E1 = in_sizes[1] / 2;
    const int NT = in_sizes[3] / DFEAT;
    const int ET = in_sizes[4] / 2;
    const long long Nmax = (N1 > NT) ? N1 : NT;
    const size_t ND = (size_t)Nmax * DFEAT;
    const size_t DD = (size_t)DFEAT * DFEAT;

    // workspace carve
    char* p = (char*)d_ws;
    bf16_t* xbf    = (bf16_t*)p; p += alup(ND * sizeof(bf16_t));
    float*  f32buf = (float*)p;  p += alup(ND * sizeof(float));
    bf16_t* aggbf  = (bf16_t*)p; p += alup(ND * sizeof(bf16_t));
    bf16_t* hbf    = (bf16_t*)p; p += alup(ND * sizeof(bf16_t));
    float*  cnt    = (float*)p;  p += alup(Nmax * sizeof(float));
    bf16_t* wl1b   = (bf16_t*)p; p += alup(DD * sizeof(bf16_t));
    bf16_t* wr1b   = (bf16_t*)p; p += alup(DD * sizeof(bf16_t));
    bf16_t* wl2b   = (bf16_t*)p; p += alup(DD * sizeof(bf16_t));
    bf16_t* wr2b   = (bf16_t*)p; p += alup(DD * sizeof(bf16_t));
    float*  emb1   = (float*)p;  p += alup(DFEAT * sizeof(float));
    float*  emb2   = (float*)p;  p += alup(DFEAT * sizeof(float));
    float*  embt   = (float*)p;  p += alup(DFEAT * sizeof(float));
    (void)ws_size; (void)n_in; (void)out_size;

    // convert all weights to bf16 once (deterministic each call)
    k_cvt_bf16<<<nblk(DD, 256), 256, 0, stream>>>(Wl1, wl1b, DD);
    k_cvt_bf16<<<nblk(DD, 256), 256, 0, stream>>>(Wr1, wr1b, DD);
    k_cvt_bf16<<<nblk(DD, 256), 256, 0, stream>>>(Wl2, wl2b, DD);
    k_cvt_bf16<<<nblk(DD, 256), 256, 0, stream>>>(Wr2, wr2b, DD);

    // graph 1 (also extracts emb2 = e1[var_idx])
    process_graph(x1, ei1, N1, E1, wl1b, wr1b, bl1, wl2b, wr2b, bl2,
                  xbf, f32buf, aggbf, hbf, cnt, emb1, vidx, emb2, stream);
    // graph t
    process_graph(xt, eit, NT, ET, wl1b, wr1b, bl1, wl2b, wr2b, bl2,
                  xbf, f32buf, aggbf, hbf, cnt, embt, vidx, nullptr, stream);

    // readout / loss -> d_out = [loss, score, label, out[768]]
    k_finalize<<<1, 256, 0, stream>>>(emb1, emb2, embt, linW, linb, Wo, ov,
                                      opn, lab, (float*)d_out);
}
